// KANLinear_44564580663829
// MI455X (gfx1250) — matrix operand order repaired and checked by reading, not compiled, not run on hardware
//
#include <hip/hip_runtime.h>
#include <hip/hip_bf16.h>

typedef __attribute__((ext_vector_type(16))) __bf16 v16bf;
typedef __attribute__((ext_vector_type(8)))  float  v8f;
typedef int v4i __attribute__((vector_size(16)));   // matches builtin param type

#define B_ROWS 4096
#define IN_F   1024
#define OUT_F  1024
#define KTOT   (IN_F * 9)      // 9216 (silu + 8 spline bases per input feature)
#define KBLKS  (KTOT / 32)     // 288 k-steps of 32
#define MTILES (B_ROWS / 16)   // 256
#define NTILES (OUT_F / 16)    // 64

#if __has_builtin(__builtin_amdgcn_global_load_async_to_lds_b128)
#define HAVE_ASYNC_LDS 1
#else
#define HAVE_ASYNC_LDS 0
#endif

// ---------------------------------------------------------------------------
// Kernel 1: pack silu(x) and the 8 cubic B-spline bases into bf16, laid out
// directly in the WMMA A-fragment layout:
//   element index = ((mtile*KBLKS + kblk)*32 + lane)*16 + j
//   lane = g*16 + (b%16);  g/j from k%32 per ISA 16-bit A 16x32 table:
//     kin 0..7  -> g=0 j=kin      kin 8..15  -> g=1 j=kin-8
//     kin 16..23-> g=0 j=8+kin-16 kin 24..31 -> g=1 j=8+kin-24
// ---------------------------------------------------------------------------
__global__ __launch_bounds__(256) void kan_pack_act(const float* __restrict__ x,
                                                    const float* __restrict__ grid,
                                                    __hip_bfloat16* __restrict__ Ap) {
    int idx = blockIdx.x * 256 + threadIdx.x;       // [0, 4096*1024)
    int b = idx >> 10;
    int i = idx & 1023;
    float xv = x[idx];

    float g[12];
#pragma unroll
    for (int j = 0; j < 12; ++j) g[j] = grid[j];

    // Cox-de-Boor, matching the reference exactly (incl. +1e-8 denominators)
    float bas[11];
#pragma unroll
    for (int j = 0; j < 11; ++j)
        bas[j] = (xv >= g[j] && xv < g[j + 1]) ? 1.0f : 0.0f;
#pragma unroll
    for (int k = 1; k <= 3; ++k) {
#pragma unroll
        for (int j = 0; j <= 10 - k; ++j) {
            float left  = (xv - g[j]) / (g[j + k] - g[j] + 1e-8f);
            float right = (g[j + k + 1] - xv) / (g[j + k + 1] - g[j + 1] + 1e-8f);
            bas[j] = left * bas[j] + right * bas[j + 1];
        }
    }

    float vals[9];
    vals[0] = xv / (1.0f + __expf(-xv));            // silu
#pragma unroll
    for (int q = 0; q < 8; ++q) vals[1 + q] = bas[q];

    int mtile = b >> 4;
    int rm    = b & 15;
#pragma unroll
    for (int q = 0; q < 9; ++q) {
        int k    = i * 9 + q;
        int kblk = k >> 5;
        int kin  = k & 31;
        int gg   = (kin < 16) ? (kin >> 3) : ((kin - 16) >> 3);
        int jj   = (kin < 16) ? (kin & 7)  : (8 + ((kin - 16) & 7));
        int lane = gg * 16 + rm;
        size_t off = ((size_t)(mtile * KBLKS + kblk) * 32 + lane) * 16 + jj;
        Ap[off] = __float2bfloat16(vals[q]);
    }
}

// ---------------------------------------------------------------------------
// Kernel 2: pack [base_weight | spline_weight] into bf16 in the WMMA
// B-fragment layout (B is 32x16 per tile; lane = g*16 + (o%16), g/j from
// kin per the same 16x32 table as A):
//   element index = ((ntile*KBLKS + kblk)*32 + lane)*16 + j
// ---------------------------------------------------------------------------
__global__ __launch_bounds__(256) void kan_pack_w(const float* __restrict__ bw,
                                                  const float* __restrict__ sw,
                                                  __hip_bfloat16* __restrict__ Wp) {
    int idx = blockIdx.x * 256 + threadIdx.x;       // [0, 1024*1024)
    int o = idx >> 10;
    int i = idx & 1023;
    int ntile = o >> 4;
    int rm    = o & 15;

    float vals[9];
    vals[0] = bw[(size_t)o * IN_F + i];
#pragma unroll
    for (int q = 0; q < 8; ++q)
        vals[1 + q] = sw[((size_t)o * IN_F + i) * 8 + q];

#pragma unroll
    for (int q = 0; q < 9; ++q) {
        int k    = i * 9 + q;
        int kblk = k >> 5;
        int kin  = k & 31;
        int gg   = (kin < 16) ? (kin >> 3) : ((kin - 16) >> 3);
        int jj   = (kin < 16) ? (kin & 7)  : (8 + ((kin - 16) & 7));
        int lane = gg * 16 + rm;
        size_t off = ((size_t)(ntile * KBLKS + kblk) * 32 + lane) * 16 + jj;
        Wp[off] = __float2bfloat16(vals[q]);
    }
}

// ---------------------------------------------------------------------------
// 16B copy global -> LDS: async (bypasses VGPRs, tracked by ASYNCcnt) when
// the builtin exists, else plain load/store.
// ---------------------------------------------------------------------------
__device__ __forceinline__ void copy16(const uint4* g, uint4* l) {
#if HAVE_ASYNC_LDS
    __builtin_amdgcn_global_load_async_to_lds_b128(
        (__attribute__((address_space(1))) v4i*)g,
        (__attribute__((address_space(3))) v4i*)l,
        /*offset=*/0, /*cpol=*/0);
#else
    *l = *g;
#endif
}

__device__ __forceinline__ void wait_async_le4() {
#if __has_builtin(__builtin_amdgcn_s_wait_asynccnt)
    __builtin_amdgcn_s_wait_asynccnt(4);
#else
    asm volatile("s_wait_asynccnt 0x4" ::: "memory");
#endif
}

// ---------------------------------------------------------------------------
// Kernel 3: GEMM  [4096 x 9216] x [9216 x 1024] -> [4096 x 1024] fp32
// 256 threads (8 wave32) per WG, 128x128 output tile, waves in 4(M) x 2(N),
// each wave owns 2x4 = 8 WMMA 16x16 accumulators. Triple-buffered LDS with
// async global->LDS copies: tile kb+2 is in flight while computing tile kb;
// s_wait_asynccnt <=4 before the barrier guarantees tile kb+1 has landed
// (each tile = 4 async b128 ops per lane; async loads complete in order).
// ---------------------------------------------------------------------------
__global__ __launch_bounds__(256) void kan_gemm(const uint4* __restrict__ Ap,
                                                const uint4* __restrict__ Wp,
                                                float* __restrict__ out) {
    // one tile chunk (one 16x32 frag block) = 512 bf16 = 64 uint4; 8 chunks/tile
    __shared__ __align__(32) uint4 sA[3][8 * 64];
    __shared__ __align__(32) uint4 sB[3][8 * 64];

    const int tid  = threadIdx.x;
    const int wave = tid >> 5;
    const int lane = tid & 31;
    const int wm   = wave & 3;          // M-direction wave index (0..3)
    const int wn   = wave >> 2;         // N-direction wave index (0..1)
    const int mtile0 = blockIdx.x * 8;  // 8 M-tiles per WG
    const int ntile0 = blockIdx.y * 8;  // 8 N-tiles per WG

    // copy assignment: one 32-thread group per chunk, 32B per thread
    const int cchunk = tid >> 5;              // 0..7
    const int cvec   = (tid & 31) * 2;        // uint4 offset within chunk

    v8f acc[2][4];
#pragma unroll
    for (int mt = 0; mt < 2; ++mt)
#pragma unroll
        for (int nt = 0; nt < 4; ++nt)
#pragma unroll
            for (int r = 0; r < 8; ++r) acc[mt][nt][r] = 0.0f;

    auto copy_tile = [&](int kb, int buf) {
        const uint4* sa = Ap + ((size_t)(mtile0 + cchunk) * KBLKS + kb) * 64 + cvec;
        const uint4* sb = Wp + ((size_t)(ntile0 + cchunk) * KBLKS + kb) * 64 + cvec;
        copy16(sa,     &sA[buf][cchunk * 64 + cvec]);
        copy16(sa + 1, &sA[buf][cchunk * 64 + cvec + 1]);
        copy16(sb,     &sB[buf][cchunk * 64 + cvec]);
        copy16(sb + 1, &sB[buf][cchunk * 64 + cvec + 1]);
    };

    // prologue: two tiles in flight
    copy_tile(0, 0);
    copy_tile(1, 1);
    wait_async_le4();                  // tile 0 landed (<=4 outstanding = tile 1)
    __syncthreads();

    for (int kb = 0; kb < KBLKS; ++kb) {
        const int cur = kb % 3;

        if (kb + 2 < KBLKS)
            copy_tile(kb + 2, (kb + 2) % 3);
        if (kb + 3 < KBLKS) {          // light L2 prefetch one tile deeper
            __builtin_prefetch(Ap + ((size_t)(mtile0 + cchunk) * KBLKS + (kb + 3)) * 64 + cvec, 0, 1);
            __builtin_prefetch(Wp + ((size_t)(ntile0 + cchunk) * KBLKS + (kb + 3)) * 64 + cvec, 0, 1);
        }

        // fragment loads from LDS (32B per lane per fragment -> 2x ds_load_b128)
        v16bf afrag[2], bfrag[4];
#pragma unroll
        for (int mt = 0; mt < 2; ++mt)
            afrag[mt] = *reinterpret_cast<const v16bf*>(&sA[cur][(wm * 2 + mt) * 64 + lane * 2]);
#pragma unroll
        for (int nt = 0; nt < 4; ++nt)
            bfrag[nt] = *reinterpret_cast<const v16bf*>(&sB[cur][(wn * 4 + nt) * 64 + lane * 2]);

#pragma unroll
        for (int mt = 0; mt < 2; ++mt)
#pragma unroll
            for (int nt = 0; nt < 4; ++nt)
                acc[mt][nt] = __builtin_amdgcn_wmma_f32_16x16x32_bf16(
                    /*neg_a=*/false, afrag[mt], /*neg_b=*/false, bfrag[nt],
                    /*c_mod=*/(short)0, acc[mt][nt],
                    /*reuse_a=*/false, /*reuse_b=*/false);

        wait_async_le4();              // tile kb+1 landed (<=4 outstanding = tile kb+2)
        __syncthreads();
    }

    // epilogue: C layout (16x16 f32, 8 VGPRs): lanes 0-15 -> M=r, lanes 16-31 -> M=8+r
    const int lg = lane >> 4;
    const int lm = lane & 15;
#pragma unroll
    for (int mt = 0; mt < 2; ++mt) {
        int row0 = (mtile0 + wm * 2 + mt) * 16 + lg * 8;
#pragma unroll
        for (int nt = 0; nt < 4; ++nt) {
            int col = (ntile0 + wn * 4 + nt) * 16 + lm;
#pragma unroll
            for (int r = 0; r < 8; ++r)
                out[(size_t)(row0 + r) * OUT_F + col] = acc[mt][nt][r];
        }
    }
}

// ---------------------------------------------------------------------------
extern "C" void kernel_launch(void* const* d_in, const int* in_sizes, int n_in,
                              void* d_out, int out_size, void* d_ws, size_t ws_size,
                              hipStream_t stream) {
    const float* x    = (const float*)d_in[0];   // [4096, 1024]
    const float* bw   = (const float*)d_in[1];   // [1024, 1024]
    const float* sw   = (const float*)d_in[2];   // [1024, 1024, 8]
    const float* grid = (const float*)d_in[3];   // [12]
    float* out = (float*)d_out;                  // [4096, 1024]

    const size_t apack_bytes = (size_t)MTILES * KBLKS * 32 * 16 * sizeof(__hip_bfloat16); // ~75.5 MB
    __hip_bfloat16* Ap = (__hip_bfloat16*)d_ws;
    __hip_bfloat16* Wp = (__hip_bfloat16*)((char*)d_ws + apack_bytes);                    // ~18.9 MB

    kan_pack_act<<<(B_ROWS * IN_F) / 256, 256, 0, stream>>>(x, grid, Ap);
    kan_pack_w<<<(OUT_F * IN_F) / 256, 256, 0, stream>>>(bw, sw, Wp);

    dim3 gemm_grid(MTILES / 8, NTILES / 8);      // 32 x 8 workgroups
    kan_gemm<<<gemm_grid, 256, 0, stream>>>((const uint4*)Ap, (const uint4*)Wp, out);
}
